// MambaEncoder2DScan_52080773431767
// MI455X (gfx1250) — compile-verified, matrix operand order checked by reading, not run on hardware
//
#include <hip/hip_runtime.h>

typedef __attribute__((ext_vector_type(16))) _Float16 v16h;
typedef __attribute__((ext_vector_type(8)))  float    v8f;

#define EPSN 1e-5f

__device__ __forceinline__ float actf(float x, int a) {
  if (a == 1) {  // gelu (tanh approx, jax default)
    float t = 0.7978845608028654f * (x + 0.044715f * x * x * x);
    return 0.5f * x * (1.f + tanhf(t));
  }
  if (a == 2) {  // silu
    return x / (1.f + __expf(-x));
  }
  if (a == 3) {  // softplus
    return (x > 20.f) ? x : log1pf(__expf(x));
  }
  return x;
}

// ---------------- direct 3-D convolution (VALU path) ----------------
__global__ void k_conv3d(const float* __restrict__ in, const float* __restrict__ w,
                         const float* __restrict__ b, float* __restrict__ out,
                         int Cin, int Cout, int D, int H, int W,
                         int Do, int Ho, int Wo, int K, int stride, int pad) {
  long idx = (long)blockIdx.x * blockDim.x + threadIdx.x;
  long total = (long)Cout * Do * Ho * Wo;
  if (idx >= total) return;
  int wo = (int)(idx % Wo); long t = idx / Wo;
  int ho = (int)(t % Ho); t /= Ho;
  int dz = (int)(t % Do); int co = (int)(t / Do);
  float acc = b[co];
  for (int ci = 0; ci < Cin; ++ci) {
    for (int kz = 0; kz < K; ++kz) {
      int z = dz * stride - pad + kz;
      if ((unsigned)z >= (unsigned)D) continue;
      for (int ky = 0; ky < K; ++ky) {
        int y = ho * stride - pad + ky;
        if ((unsigned)y >= (unsigned)H) continue;
        for (int kx = 0; kx < K; ++kx) {
          int x = wo * stride - pad + kx;
          if ((unsigned)x >= (unsigned)W) continue;
          acc += in[((long)ci * D + z) * H * W + (long)y * W + x] *
                 w[(((long)co * Cin + ci) * K + kz) * K * K + (long)ky * K + kx];
        }
      }
    }
  }
  out[idx] = acc;
}

// ---------------- instance norm: per-channel stats (LDS reduce) ----------------
__global__ void k_inorm_stats(const float* __restrict__ in, float* __restrict__ mean,
                              float* __restrict__ rstd, long S) {
  int c = blockIdx.x;
  const float* p = in + (long)c * S;
  float s = 0.f, s2 = 0.f;
  for (long i = threadIdx.x; i < S; i += blockDim.x) { float v = p[i]; s += v; s2 += v * v; }
  __shared__ float sh[256], sh2[256];
  sh[threadIdx.x] = s; sh2[threadIdx.x] = s2;
  __syncthreads();
  for (int o = blockDim.x / 2; o > 0; o >>= 1) {
    if ((int)threadIdx.x < o) { sh[threadIdx.x] += sh[threadIdx.x + o]; sh2[threadIdx.x] += sh2[threadIdx.x + o]; }
    __syncthreads();
  }
  if (threadIdx.x == 0) {
    float m = sh[0] / (float)S;
    float v = sh2[0] / (float)S - m * m;
    mean[c] = m; rstd[c] = rsqrtf(v + EPSN);
  }
}

__global__ void k_inorm_apply(const float* __restrict__ in, const float* __restrict__ mean,
                              const float* __restrict__ rstd, float* __restrict__ out,
                              const float* __restrict__ resid, long S, int C, int relu) {
  long idx = (long)blockIdx.x * blockDim.x + threadIdx.x;
  long total = (long)C * S;
  if (idx >= total) return;
  int c = (int)(idx / S);
  float v = (in[idx] - mean[c]) * rstd[c];
  if (relu) v = fmaxf(v, 0.f);
  if (resid) v += resid[idx];
  out[idx] = v;
}

// ---------------- layer norm over channels (one block per token row) ----------------
__global__ void k_lnorm(const float* __restrict__ in, float* __restrict__ out,
                        const float* __restrict__ g, const float* __restrict__ b, int C) {
  long row = blockIdx.x;
  const float* p = in + row * C;
  float s = 0.f, s2 = 0.f;
  for (int i = threadIdx.x; i < C; i += blockDim.x) { float v = p[i]; s += v; s2 += v * v; }
  __shared__ float sh[256], sh2[256];
  sh[threadIdx.x] = s; sh2[threadIdx.x] = s2;
  __syncthreads();
  for (int o = blockDim.x / 2; o > 0; o >>= 1) {
    if ((int)threadIdx.x < o) { sh[threadIdx.x] += sh[threadIdx.x + o]; sh2[threadIdx.x] += sh2[threadIdx.x + o]; }
    __syncthreads();
  }
  float m = sh[0] / (float)C;
  float var = sh2[0] / (float)C - m * m;
  float rs = rsqrtf(var + EPSN);
  for (int i = threadIdx.x; i < C; i += blockDim.x)
    out[row * C + i] = (p[i] - m) * rs * g[i] + b[i];
}

// ---------------- column mean over tokens: g[c] = mean_l seq[l,c] ----------------
__global__ void k_colmean(const float* __restrict__ in, float* __restrict__ out, long M, int C) {
  int c = blockIdx.x;
  float s = 0.f;
  for (long i = threadIdx.x; i < M; i += blockDim.x) s += in[i * C + c];
  __shared__ float sh[256];
  sh[threadIdx.x] = s;
  __syncthreads();
  for (int o = blockDim.x / 2; o > 0; o >>= 1) {
    if ((int)threadIdx.x < o) sh[threadIdx.x] += sh[threadIdx.x + o];
    __syncthreads();
  }
  if (threadIdx.x == 0) out[c] = sh[0] / (float)M;
}

// ---------------- WMMA GEMM: C[M,N] = act(A[M,K]*B[K,N] + bias) ----------------
// f32 in memory, f16 operands, f32 accumulation via v_wmma_f32_16x16x32_f16.
// A tile (16x32) staged once per block through LDS (8x reuse across the 8 waves);
// B interior k-steps take an unchecked fast path (wave-uniform predicate).
__global__ void __launch_bounds__(256)
k_gemm_wmma(const float* __restrict__ A, int lda, const float* __restrict__ B, int ldb,
            const float* __restrict__ bias, int bias_per_m,
            float* __restrict__ C, int ldc, int M, int N, int K, int act) {
  __shared__ _Float16 shA[16 * 32];
  int tid = threadIdx.x;
  int wave = tid >> 5;
  int lane = tid & 31;
  int tileM = blockIdx.x * 16;
  int tileN = (blockIdx.y * 8 + wave) * 16;

  v8f acc = {};
  int ncol = tileN + (lane & 15);
  int khalf = (lane >> 4) * 8;      // §7.12.2 16-bit fragment K-halves
  int arow = (lane & 15) * 32;
  bool nfull = (tileN + 16 <= N);   // wave-uniform

  for (int k0 = 0; k0 < K; k0 += 32) {
    // ---- cooperative A-tile stage: 16x32 f32 -> f16 in LDS ----
    {
      int e = tid * 2;
      int r = e >> 5, kk = e & 31;
      int gr = tileM + r, gk = k0 + kk;
      float v0 = 0.f, v1 = 0.f;
      if (gr < M) {
        if (gk < K)     v0 = A[(long)gr * lda + gk];
        if (gk + 1 < K) v1 = A[(long)gr * lda + gk + 1];
      }
      __syncthreads();   // previous iteration's fragment reads complete
      shA[e]     = (_Float16)v0;
      shA[e + 1] = (_Float16)v1;
      __syncthreads();
    }

    v16h a, b;
#pragma unroll
    for (int t = 0; t < 8; ++t) {
      a[t]     = shA[arow + khalf + t];
      a[8 + t] = shA[arow + 16 + khalf + t];
    }

    if (nfull && (k0 + 32 <= K)) {
      // interior fast path: unconditional, unguarded B loads
      const float* pb = B + (long)(k0 + khalf) * ldb + ncol;
#pragma unroll
      for (int t = 0; t < 8; ++t) {
        b[t]     = (_Float16)pb[(long)t * ldb];
        b[8 + t] = (_Float16)pb[(long)(16 + t) * ldb];
      }
    } else {
#pragma unroll
      for (int t = 0; t < 8; ++t) {
        int k  = k0 + khalf + t;
        int k2 = k + 16;
        float v0 = 0.f, v1 = 0.f;
        if (ncol < N) {
          if (k  < K) v0 = B[(long)k  * ldb + ncol];
          if (k2 < K) v1 = B[(long)k2 * ldb + ncol];
        }
        b[t] = (_Float16)v0; b[8 + t] = (_Float16)v1;
      }
    }

    if (k0 + 32 < K) {  // emits global_prefetch_b8 on gfx1250
      __builtin_prefetch(&A[(long)(tileM + (lane & 15)) * lda + k0 + 32], 0, 3);
      __builtin_prefetch(&B[(long)(k0 + 32 + khalf) * ldb + ncol], 0, 3);
    }

    acc = __builtin_amdgcn_wmma_f32_16x16x32_f16(false, a, false, b, (short)0, acc, false, false);
  }

  int nn = tileN + (lane & 15);
  if (nn >= N || tileN >= N) return;
#pragma unroll
  for (int v = 0; v < 8; ++v) {
    int mm = tileM + v + 8 * (lane >> 4);  // §7.12.2 32-bit C/D layout
    if (mm >= M) continue;
    float val = acc[v];
    if (bias) val += bias_per_m ? bias[mm] : bias[nn];
    C[(long)mm * ldc + nn] = actf(val, act);
  }
}

// ---------------- causal depthwise conv1d (k=4) + bias + silu ----------------
__global__ void k_dwconv_silu(const float* __restrict__ u, const float* __restrict__ w,
                              const float* __restrict__ b, float* __restrict__ out,
                              long M, int C) {
  long idx = (long)blockIdx.x * blockDim.x + threadIdx.x;
  if (idx >= M * C) return;
  int c = (int)(idx % C);
  long l = idx / C;
  float acc = b[c];
#pragma unroll
  for (int k = 0; k < 4; ++k) {
    long ll = l - 3 + k;
    if (ll >= 0) acc += u[ll * C + c] * w[c * 4 + k];
  }
  out[idx] = actf(acc, 2);
}

// ---------------- selective scan: wave per channel, 16 state lanes ----------------
__global__ void __launch_bounds__(32)
k_scan(const float* __restrict__ dt, const float* __restrict__ u,
       const float* __restrict__ proj, const float* __restrict__ A_log,
       const float* __restrict__ Dp, float* __restrict__ y,
       long M, int C, int PL, int R) {
  int c = blockIdx.x;
  int lane = threadIdx.x;
  bool on = lane < 16;
  float Av = on ? -__expf(A_log[(long)c * 16 + lane]) : 0.f;
  float Dv = Dp[c];
  float h = 0.f;
  for (long l = 0; l < M; ++l) {
    float dtv = dt[l * C + c];
    float uv  = u[l * C + c];
    float bm  = on ? proj[l * PL + R + lane] : 0.f;
    float cm  = on ? proj[l * PL + R + 16 + lane] : 0.f;
    h = __expf(dtv * Av) * h + dtv * uv * bm;
    float p = h * cm;
    p += __shfl_xor(p, 1, 32);
    p += __shfl_xor(p, 2, 32);
    p += __shfl_xor(p, 4, 32);
    p += __shfl_xor(p, 8, 32);
    if (lane == 0) y[l * C + c] = p + uv * Dv;
  }
}

// ---------------- small glue kernels ----------------
__global__ void k_split(const float* __restrict__ xz, float* __restrict__ u,
                        float* __restrict__ z, long M, int C) {
  long idx = (long)blockIdx.x * blockDim.x + threadIdx.x;
  if (idx >= M * C) return;
  int c = (int)(idx % C); long r = idx / C;
  u[idx] = xz[r * 2 * C + c];
  z[idx] = xz[r * 2 * C + C + c];
}
__global__ void k_mul_silu(float* __restrict__ y, const float* __restrict__ z, long n) {
  long idx = (long)blockIdx.x * blockDim.x + threadIdx.x;
  if (idx < n) y[idx] *= actf(z[idx], 2);
}
__global__ void k_add(const float* __restrict__ a, const float* __restrict__ b,
                      float* __restrict__ out, long n) {
  long idx = (long)blockIdx.x * blockDim.x + threadIdx.x;
  if (idx < n) out[idx] = a[idx] + b[idx];
}
__global__ void k_add_skip(float* __restrict__ seq, const float* __restrict__ y, long L, int C) {
  long idx = (long)blockIdx.x * blockDim.x + threadIdx.x;
  if (idx < L * C) seq[idx] += y[idx + C];  // skip global token row
}
__global__ void k_prepend(const float* __restrict__ g, const float* __restrict__ seq,
                          float* __restrict__ s, long L, int C) {
  long idx = (long)blockIdx.x * blockDim.x + threadIdx.x;
  if (idx >= (L + 1) * C) return;
  int c = (int)(idx % C); long r = idx / C;
  s[idx] = (r == 0) ? g[c] : seq[(r - 1) * C + c];
}
__global__ void k_cl2lc(const float* __restrict__ x, float* __restrict__ seq, int C, long L) {
  long idx = (long)blockIdx.x * blockDim.x + threadIdx.x;
  if (idx >= (long)C * L) return;
  int c = (int)(idx / L); long l = idx % L;
  seq[l * C + c] = x[idx];
}
__global__ void k_lc2cl(const float* __restrict__ seq, float* __restrict__ x, int C, long L) {
  long idx = (long)blockIdx.x * blockDim.x + threadIdx.x;
  if (idx >= (long)C * L) return;
  int c = (int)(idx / L); long l = idx % L;
  x[idx] = seq[l * C + c];
}
__global__ void k_copy(const float* __restrict__ in, float* __restrict__ out, long n) {
  long idx = (long)blockIdx.x * blockDim.x + threadIdx.x;
  if (idx < n) out[idx] = in[idx];
}

// =====================================================================
extern "C" void kernel_launch(void* const* d_in, const int* in_sizes, int n_in,
                              void* d_out, int out_size, void* d_ws, size_t ws_size,
                              hipStream_t stream) {
  (void)in_sizes; (void)n_in; (void)out_size; (void)ws_size;
  const int DIMS_[5] = {48, 96, 192, 384, 768};

  // Input pytree flattening (dict keys sorted, JAX convention):
  // 0: x ; 1..208: stages[0..3] x 52 ; 209: stem_b ; 210: stem_w
  auto P = [&](int i) { return (const float*)d_in[i]; };
  const float* X_IN = P(0);

  // workspace layout (floats)
  float* W0 = (float*)d_ws;
  const long MB = 1048576;
  float* X    = W0;              // 2M  current feature map [C, L]
  float* T1   = W0 + 2 * MB;     // 2M  (also UC = post-conv u)
  float* T2   = W0 + 4 * MB;     // 2M  (also YO = out-proj result)
  float* T3   = W0 + 6 * MB;     // 2M
  float* SEQ  = W0 + 8 * MB;     // 2M  tokens [L, C]
  float* SBUF = W0 + 10 * MB;    // 2M  [L+1, C] (also FFN ln2 out)
  float* XZ   = W0 + 12 * MB;    // 4M  [L+1, 2C] (also FFN hidden)
  float* PROJ = W0 + 16 * MB;    // 2M  [L+1, R+32]
  float* U    = W0 + 18 * MB;    // 2M
  float* Z    = W0 + 20 * MB;    // 2M
  float* DT   = W0 + 22 * MB;    // 2M
  float* Y    = W0 + 24 * MB;    // 2M
  float* STAT = W0 + 26 * MB;    // mean
  float* RSTD = STAT + 2048;
  float* G    = STAT + 4096;

  auto blocks1 = [](long n) { return dim3((unsigned)((n + 255) / 256)); };
  auto conv = [&](const float* in, const float* w, const float* b, float* out,
                  int Cin, int Cout, int Din, int K, int stride, int pad, int Dout) {
    long total = (long)Cout * Dout * Dout * Dout;
    k_conv3d<<<blocks1(total), 256, 0, stream>>>(in, w, b, out, Cin, Cout,
                                                 Din, Din, Din, Dout, Dout, Dout, K, stride, pad);
  };
  auto inorm = [&](const float* in, float* out, const float* resid, long S, int C, int relu) {
    k_inorm_stats<<<C, 256, 0, stream>>>(in, STAT, RSTD, S);
    k_inorm_apply<<<blocks1((long)C * S), 256, 0, stream>>>(in, STAT, RSTD, out, resid, S, C, relu);
  };
  auto gemm = [&](const float* A, int lda, const float* B, int ldb, const float* bias,
                  int bias_m, float* C, int ldc, long M, int N, int K, int act) {
    dim3 g((unsigned)((M + 15) / 16), (unsigned)((N + 127) / 128));
    k_gemm_wmma<<<g, 256, 0, stream>>>(A, lda, B, ldb, bias, bias_m, C, ldc, (int)M, N, K, act);
  };

  float* OUT = (float*)d_out;
  long outOff = 0;

  // --- stem: conv k7 s2 p3, 1 -> 48, 64^3 -> 32^3 ---
  conv(X_IN, P(210), P(209), X, 1, 48, 64, 7, 2, 3, 32);
  {
    long n = 48L * 32 * 32 * 32;
    k_copy<<<blocks1(n), 256, 0, stream>>>(X, OUT, n);
    outOff += n;
  }

  int sd = 32;
  for (int i = 0; i < 4; ++i) {
    int d = DIMS_[i], dn = DIMS_[i + 1];
    int R = d / 16; if (R < 1) R = 1;
    int PL = R + 32;
    long L = (long)sd * sd * sd, Ls = L + 1;
    int sb = 1 + 52 * i;
    const float* pe_b = P(sb + 38); const float* pe_g = P(sb + 39);
    const float* down_b = P(sb + 40); const float* down_w = P(sb + 41);
    const float* b1 = P(sb + 42); const float* b2 = P(sb + 43); const float* b3 = P(sb + 44);
    const float* w1 = P(sb + 45); const float* w2 = P(sb + 46); const float* w3 = P(sb + 47);
    const float* mlp1_b = P(sb + 48); const float* mlp1_w = P(sb + 49);
    const float* mlp2_b = P(sb + 50); const float* mlp2_w = P(sb + 51);

    // ---- GSC ----
    conv(X, w1, b1, T1, d, d, sd, 3, 1, 1, sd);
    inorm(T1, T1, nullptr, L, d, 1);
    conv(X, w2, b2, T2, d, d, sd, 1, 1, 0, sd);
    inorm(T2, T2, nullptr, L, d, 1);
    k_add<<<blocks1((long)d * L), 256, 0, stream>>>(T1, T2, T2, (long)d * L);
    conv(T2, w3, b3, T3, d, d, sd, 3, 1, 1, sd);
    inorm(T3, X, X, L, d, 1);  // x = x + relu(inorm(conv(x1+x2)))

    // ---- ASSB ----
    k_cl2lc<<<blocks1((long)d * L), 256, 0, stream>>>(X, SEQ, d, L);
    k_lnorm<<<(unsigned)L, 256, 0, stream>>>(SEQ, SEQ, pe_g, pe_b, d);
    k_colmean<<<d, 256, 0, stream>>>(SEQ, G, L, d);

    for (int blk = 0; blk < 2; ++blk) {
      int bb = sb + 19 * blk;
      const float* fc1_b = P(bb + 0); const float* fc1_w = P(bb + 1);
      const float* fc2_b = P(bb + 2); const float* fc2_w = P(bb + 3);
      const float* ln1_b = P(bb + 4); const float* ln1_g = P(bb + 5);
      const float* ln2_b = P(bb + 6); const float* ln2_g = P(bb + 7);
      const float* A_log = P(bb + 8);  const float* Dp = P(bb + 9);
      const float* conv_b = P(bb + 10); const float* conv_w = P(bb + 11);
      const float* dt_b = P(bb + 12);  const float* dt_w = P(bb + 13);
      const float* in_b = P(bb + 14);  const float* in_w = P(bb + 15);
      const float* out_b = P(bb + 16); const float* out_w = P(bb + 17);
      const float* xproj_w = P(bb + 18);

      k_prepend<<<blocks1(Ls * d), 256, 0, stream>>>(G, SEQ, SBUF, L, d);
      k_lnorm<<<(unsigned)Ls, 256, 0, stream>>>(SBUF, SBUF, ln1_g, ln1_b, d);
      // s6:
      gemm(SBUF, d, in_w, 2 * d, in_b, 0, XZ, 2 * d, Ls, 2 * d, d, 0);
      k_split<<<blocks1(Ls * d), 256, 0, stream>>>(XZ, U, Z, Ls, d);
      k_dwconv_silu<<<blocks1(Ls * d), 256, 0, stream>>>(U, conv_w, conv_b, T1, Ls, d);
      gemm(T1, d, xproj_w, PL, nullptr, 0, PROJ, PL, Ls, PL, d, 0);
      gemm(PROJ, PL, dt_w, d, dt_b, 0, DT, d, Ls, d, R, 3);           // softplus
      k_scan<<<d, 32, 0, stream>>>(DT, T1, PROJ, A_log, Dp, Y, Ls, d, PL, R);
      k_mul_silu<<<blocks1(Ls * d), 256, 0, stream>>>(Y, Z, Ls * d);
      gemm(Y, d, out_w, d, out_b, 0, T2, d, Ls, d, d, 0);
      k_add_skip<<<blocks1(L * d), 256, 0, stream>>>(SEQ, T2, L, d);
      // FFN:
      k_lnorm<<<(unsigned)L, 256, 0, stream>>>(SEQ, SBUF, ln2_g, ln2_b, d);
      gemm(SBUF, d, fc1_w, 2 * d, fc1_b, 0, XZ, 2 * d, L, 2 * d, d, 1);  // gelu
      gemm(XZ, 2 * d, fc2_w, d, fc2_b, 0, Y, d, L, d, 2 * d, 0);
      k_add<<<blocks1(L * d), 256, 0, stream>>>(SEQ, Y, SEQ, L * d);
    }
    k_lc2cl<<<blocks1((long)d * L), 256, 0, stream>>>(SEQ, X, d, L);

    // ---- down + channel MLP ----
    inorm(X, T3, nullptr, L, d, 0);
    int sd2 = sd / 2;
    long S2 = (long)sd2 * sd2 * sd2;
    conv(T3, down_w, down_b, T1, d, dn, sd, 2, 2, 0, sd2);
    inorm(T1, T2, nullptr, S2, dn, 0);
    gemm(mlp1_w, dn, T2, (int)S2, mlp1_b, 1, T3, (int)S2, 2 * dn, (int)S2, dn, 1);       // gelu
    gemm(mlp2_w, 2 * dn, T3, (int)S2, mlp2_b, 1, OUT + outOff, (int)S2, dn, (int)S2, 2 * dn, 0);
    outOff += (long)dn * S2;

    k_copy<<<blocks1((long)dn * S2), 256, 0, stream>>>(T1, X, (long)dn * S2);
    sd = sd2;
  }
}
